// VARImputer_61933428408497
// MI455X (gfx1250) — compile-verified, matrix operand order checked by reading, not run on hardware
//
#include <hip/hip_runtime.h>

typedef __attribute__((ext_vector_type(16))) _Float16 v16h;
typedef __attribute__((ext_vector_type(8)))  _Float16 v8h;
typedef __attribute__((ext_vector_type(8)))  float    v8f;

#define VORDER 5
#define NSLOT  6                   // ring slots = ORDER+1 -> 1 barrier / step
#define B_TOT  256
#define S_LEN  2048
#define F_DIM  64
#define K_DIM  (VORDER * F_DIM)    // 320
#define TILE_M 16
#define FRAME  (TILE_M * F_DIM)    // 1024 f16 elements per frame

#if defined(__has_builtin)
#  if __has_builtin(__builtin_amdgcn_sched_group_barrier)
#    define SCHED_GROUP(mask, cnt, id) __builtin_amdgcn_sched_group_barrier(mask, cnt, id)
#  endif
#endif
#ifndef SCHED_GROUP
#  define SCHED_GROUP(mask, cnt, id)
#endif

// ---------------- Kernel 1: masked mean over S per (b, f) ----------------
__global__ __launch_bounds__(256)
void var_mean_kernel(const float* __restrict__ x,
                     const int*   __restrict__ mask,
                     float*       __restrict__ mean) {
  const int b   = blockIdx.x;
  const int tid = threadIdx.x;
  const int f   = tid & 63;
  const int sy  = tid >> 6;               // 0..3
  const float* xp = x    + (size_t)b * S_LEN * F_DIM;
  const int*   mp = mask + (size_t)b * S_LEN * F_DIM;
  float s = 0.f, c = 0.f;
  for (int t = sy; t < S_LEN; t += 4) {
    const size_t idx = (size_t)t * F_DIM + f;
    const int m = mp[idx];
    s += m ? xp[idx] : 0.f;              // x * mask
    c += (float)m;
  }
  __shared__ float sm[4][64];
  __shared__ float cm[4][64];
  sm[sy][f] = s;
  cm[sy][f] = c;
  __syncthreads();
  if (sy == 0) {
    const float st = sm[0][f] + sm[1][f] + sm[2][f] + sm[3][f];
    const float ct = cm[0][f] + cm[1][f] + cm[2][f] + cm[3][f];
    mean[b * F_DIM + f] = st / (ct + 1e-8f);
  }
}

// ---------------- Kernel 2: sequential VAR imputation scan ----------------
// 16 blocks x 128 threads (4 waves). Block owns 16 batch rows; wave w owns
// output features [16w, 16w+16). Window lives in LDS as a 6-slot ring of
// 16x64 f16 frames; W^T lives in VGPRs as 10 f16 WMMA B-fragments per wave.
// Slot map at step t: frame of age a (a=0 oldest) was written at step t-5+a
// -> slot (t+a+1) % 6; step t's new frame -> slot t % 6 (disjoint from all
// read slots, so a single end-of-step barrier is sufficient).
// Output stores for step t are deferred to the top of step t+1 so their L2
// ack latency hides under the next WMMA chain instead of the barrier wait.
__global__ __launch_bounds__(128)
void var_impute_kernel(const float* __restrict__ x,
                       const int*   __restrict__ mask,
                       const float* __restrict__ W,
                       const float* __restrict__ bias,
                       const float* __restrict__ mean,
                       float*       __restrict__ out) {
  __shared__ _Float16 win[NSLOT * FRAME];   // 6 * 16 * 64 * 2B = 12 KB

  const int tid  = threadIdx.x;
  const int lane = tid & 31;
  const int wv   = tid >> 5;               // wave id 0..3
  const int n0   = wv * 16;                // output feature tile base
  const int b0   = blockIdx.x * TILE_M;    // batch tile base

  // ---- Pack this wave's W^T slice into 10 WMMA B fragments (f16) ----
  // Lane L holds column n = n0 + (L&15); K runs {ks..ks+7} and {ks+16..ks+23}
  // of each 32-wide K chunk, ks = (L>>4)*8 (mirrors the documented A layout).
  v16h wf[10];
  {
    const int n  = n0 + (lane & 15);
    const int ks = (lane >> 4) * 8;
    const float* wr = W + (size_t)n * K_DIM;   // row n of W == column n of W^T
#pragma unroll
    for (int c = 0; c < 10; ++c) {
      const int kb = c * 32 + ks;
#pragma unroll
      for (int j = 0; j < 8; ++j) {
        wf[c][j]     = (_Float16)wr[kb + j];
        wf[c][8 + j] = (_Float16)wr[kb + 16 + j];
      }
    }
  }
  const float bv = bias[n0 + (lane & 15)];   // bias for this lane's feature

  // ---- Init: ages 0..4 at t=0 were "written" at steps -5..-1 -> slots 1..5
  for (int i = tid; i < FRAME; i += 128) {
    const int m = i >> 6, f = i & 63;
    const _Float16 h = (_Float16)mean[(b0 + m) * F_DIM + f];
#pragma unroll
    for (int p = 1; p < NSLOT; ++p) win[p * FRAME + i] = h;
  }
  __syncthreads();

  // ---- Per-lane address precomputation ----
  const int    mlo       = (lane >> 4) * 8;        // D-layout: m = mlo + r
  const int    fme       = n0 + (lane & 15);       // D-layout: lane's feature
  const size_t rowStride = (size_t)S_LEN * F_DIM;
  const size_t base0     = (size_t)(b0 + mlo) * rowStride + fme;

  const int ldsA_row  = (lane & 15) * F_DIM;       // A-layout: m = lane&15
  const int ldsA_col0 = (lane >> 4) * 8;           // hi half-wave starts +8
  const int ldsD_base = mlo * F_DIM + fme;

  float vprev[8];                                  // deferred out values
#pragma unroll
  for (int r = 0; r < 8; ++r) vprev[r] = 0.f;

  // ---- Sequential scan, unrolled by NSLOT so ring slots are static ----
  for (int t0 = 0; t0 < S_LEN; t0 += NSLOT) {
#pragma unroll
    for (int ph = 0; ph < NSLOT; ++ph) {
      const int t = t0 + ph;                       // block-uniform

      // 0) deferred global stores of step t-1 (post-barrier; completes
      //    under this step's chain, off the barrier's storecnt wait)
      if (t >= 1 && t <= S_LEN) {
        const size_t poff = (size_t)(t - 1) * F_DIM;
#pragma unroll
        for (int r = 0; r < 8; ++r)
          out[base0 + (size_t)r * rowStride + poff] = vprev[r];
      }

      if (t < S_LEN) {
        const size_t toff = (size_t)t * F_DIM;

        // 1) issue all global x/mask loads up front (hide under WMMA chain)
        float xo[8], mf[8];
#pragma unroll
        for (int r = 0; r < 8; ++r) {
          const size_t gi = base0 + (size_t)r * rowStride + toff;
          xo[r] = x[gi];
          mf[r] = (float)mask[gi];
        }

        // 2) fetch all 10 A fragments from the LDS ring
        v16h a[10];
#pragma unroll
        for (int c = 0; c < 10; ++c) {
          const int slot = (ph + (c >> 1) + 1) % NSLOT;  // compile-time
          const int off  = slot * FRAME + ldsA_row + (c & 1) * 32 + ldsA_col0;
          const v8h lo = *(const v8h*)&win[off];          // K: ks..ks+7
          const v8h hi = *(const v8h*)&win[off + 16];     // K: ks+16..ks+23
          a[c] = __builtin_shufflevector(lo, hi,
                     0,1,2,3,4,5,6,7,8,9,10,11,12,13,14,15);
        }

        // 3) x_hat tile: 10 chained WMMAs over K=320, C seeded with bias
        v8f acc;
#pragma unroll
        for (int r = 0; r < 8; ++r) acc[r] = bv;
#pragma unroll
        for (int c = 0; c < 10; ++c)
          acc = __builtin_amdgcn_wmma_f32_16x16x32_f16(
                    false, a[c], false, wf[c], (short)0, acc, false, false);

        // pipeline shape: all vmem reads, then all ds reads, then all wmma
        SCHED_GROUP(0x020, 16, 0);   // 16 VMEM reads (x/mask)
        SCHED_GROUP(0x100, 20, 0);   // 20 DS reads (A fragments)
        SCHED_GROUP(0x008, 10, 0);   // 10 WMMA

        // 4) branch-free merge; keep out values for deferred store;
        //    write new ring frame (f16, slot ph)
#pragma unroll
        for (int r = 0; r < 8; ++r) {
          const float v = fmaf(mf[r], xo[r] - acc[r], acc[r]);  // m? x : x_hat
          vprev[r] = v;
          win[ph * FRAME + r * F_DIM + ldsD_base] = (_Float16)v;
        }
      }
      __syncthreads();   // new frame visible before next step's A loads
    }
  }
}

extern "C" void kernel_launch(void* const* d_in, const int* in_sizes, int n_in,
                              void* d_out, int out_size, void* d_ws, size_t ws_size,
                              hipStream_t stream) {
  const float* x    = (const float*)d_in[0];
  const int*   mask = (const int*)  d_in[1];
  const float* W    = (const float*)d_in[2];
  const float* bias = (const float*)d_in[3];
  float*       out  = (float*)d_out;
  float*       mean = (float*)d_ws;          // B_TOT * F_DIM floats = 64 KB

  var_mean_kernel<<<B_TOT, 256, 0, stream>>>(x, mask, mean);
  var_impute_kernel<<<B_TOT / TILE_M, 128, 0, stream>>>(x, mask, W, bias, mean, out);
}
